// MambaBlock_6966436954174
// MI455X (gfx1250) — matrix operand order repaired and checked by reading, not compiled, not run on hardware
//
#include <hip/hip_runtime.h>
#include <hip/hip_bf16.h>

typedef __attribute__((ext_vector_type(16))) __bf16 v16bf;
typedef __attribute__((ext_vector_type(8)))  float  v8f;

#define BM 128
#define BN 128
#define BK 32
#define LDSS 40   // ushort stride (32 + 8 pad), even -> 4B-aligned pair reads

union FragBF { v16bf bf; unsigned int u32[8]; };

__device__ __forceinline__ unsigned short f2bf(float f) {
    union { float f; unsigned int u; } c; c.f = f;
    unsigned int u = c.u;
    u += 0x7FFFu + ((u >> 16) & 1u);   // round-to-nearest-even
    return (unsigned short)(u >> 16);
}

__global__ void cvt_bf16_kernel(const float* __restrict__ src,
                                unsigned short* __restrict__ dst, int n) {
    int i = blockIdx.x * 256 + threadIdx.x;
    if (i < n) dst[i] = f2bf(src[i]);
}

// Convert with zero-padding of the trailing dim: src (rows x scol) fp32 ->
// dst (rows x dcol) bf16, cols >= scol zero-filled.  Lets the N=96 GEMM run
// with a guard-free 128-wide tile.
__global__ void cvt_bf16_pad_kernel(const float* __restrict__ src,
                                    unsigned short* __restrict__ dst,
                                    int rows, int scol, int dcol) {
    int i = blockIdx.x * 256 + threadIdx.x;
    if (i >= rows * dcol) return;
    int r = i / dcol, c = i - r * dcol;
    dst[i] = (c < scol) ? f2bf(src[r * scol + c]) : (unsigned short)0;
}

// One block per row of 1024; emits bf16 normalized hidden.
__global__ __launch_bounds__(256)
void rmsnorm_kernel(const float* __restrict__ x, const float* __restrict__ w,
                    unsigned short* __restrict__ out) {
    __shared__ float red[256];
    const int row = blockIdx.x;
    const float* xr = x + (size_t)row * 1024;
    float vals[4];
    float s = 0.f;
    #pragma unroll
    for (int i = 0; i < 4; i++) {
        vals[i] = xr[threadIdx.x + i * 256];
        s += vals[i] * vals[i];
    }
    red[threadIdx.x] = s;
    __syncthreads();
    for (int off = 128; off > 0; off >>= 1) {
        if (threadIdx.x < (unsigned)off) red[threadIdx.x] += red[threadIdx.x + off];
        __syncthreads();
    }
    const float scale = rsqrtf(red[0] * (1.0f / 1024.0f) + 1e-5f);
    #pragma unroll
    for (int i = 0; i < 4; i++) {
        int c = threadIdx.x + i * 256;
        out[(size_t)row * 1024 + c] = f2bf(vals[i] * scale * w[c]);
    }
}

// Compile-time-shaped bf16 GEMM: C(Mc x Nc) = A(Mc x Kc) * B(Kc x Nc), both
// row-major bf16.  Mc % 128 == 0, Nc % 128 == 0, Kc % 32 == 0 and all powers
// of two -> staging loops are guard-free and the next-tile prefetch is
// branchless (wraps with & (Kc-1)).
// MODE 0: Nc=4096 split -> out0 = x cols[0,2048), out1 = silu(z) cols[2048,4096)
// MODE 1: Nc=128 (B zero-padded from 96): out0 = x_dbl fp32 (ld 96, gn<96);
//         cols<64 also -> out_bf (dt A-matrix, ld 64)
// MODE 2: out0 = softplus(v + extra[n]) fp32 (ld 2048)
// MODE 3: out0 = v + extra[row*1024+n] (residual add, ld 1024)
template<int MODE, int Mc, int Nc, int Kc>
__global__ __launch_bounds__(256)
void gemm_bf16_kernel(const unsigned short* __restrict__ A,
                      const unsigned short* __restrict__ B,
                      float* __restrict__ out0,
                      float* __restrict__ out1,
                      unsigned short* __restrict__ out_bf,
                      const float* __restrict__ extra) {
    __shared__ unsigned short lds_a[BM * LDSS];  // [m][k]
    __shared__ unsigned short lds_b[BN * LDSS];  // [n][k] (transposed on fill)

    const int tid  = threadIdx.x;
    const int lane = tid & 31;
    const int wave = tid >> 5;
    const int wm   = wave & 3;     // 4 waves along M (32 rows each)
    const int wn   = wave >> 2;    // 2 waves along N (64 cols each)
    const int m0   = blockIdx.y * BM;
    const int n0   = blockIdx.x * BN;

    const v8f vzero = {0.f,0.f,0.f,0.f,0.f,0.f,0.f,0.f};
    v8f acc[2][4];
    #pragma unroll
    for (int i = 0; i < 2; i++)
        #pragma unroll
        for (int j = 0; j < 4; j++) acc[i][j] = vzero;

    const int mrow = lane & 15;
    const int half = lane >> 4;

    // Per-thread staging coordinates (constant across K loop).
    const int a_row = tid >> 4;
    const int a_cp  = tid & 15;
    const int b_kr  = tid >> 6;
    const int b_np  = tid & 63;

    for (int k0 = 0; k0 < Kc; k0 += BK) {
        // Branchless next-K-tile hint (Kc is a power of two; last iter wraps
        // to tile 0 which is already hot).  Emits global_prefetch_b8.
        const int kn = (k0 + BK) & (Kc - 1);
        __builtin_prefetch(A + (size_t)(m0 + a_row) * Kc + kn + a_cp * 2, 0, 1);
        __builtin_prefetch(B + (size_t)(kn + b_kr) * Nc + n0 + b_np * 2, 0, 1);

        // ---- stage A tile 128x32 (2048 dword loads, coalesced, no guards) ----
        #pragma unroll
        for (int i = 0; i < 8; i++) {
            int u   = tid + i * 256;
            int row = u >> 4;       // 16 dwords per row
            int cp  = u & 15;
            unsigned int val =
                *(const unsigned int*)(A + (size_t)(m0 + row) * Kc + k0 + cp * 2);
            *(unsigned int*)(&lds_a[row * LDSS + cp * 2]) = val;
        }
        // ---- stage B tile 32x128, transpose to [n][k], no guards ----
        #pragma unroll
        for (int i = 0; i < 8; i++) {
            int u  = tid + i * 256;
            int kr = u >> 6;        // 64 dwords per K-row
            int np = u & 63;
            unsigned int val =
                *(const unsigned int*)(B + (size_t)(k0 + kr) * Nc + n0 + np * 2);
            lds_b[(np * 2)     * LDSS + kr] = (unsigned short)(val & 0xFFFFu);
            lds_b[(np * 2 + 1) * LDSS + kr] = (unsigned short)(val >> 16);
        }
        __syncthreads();

        // ---- fragments per ISA 7.12.2 layouts ----
        FragBF afr[2], bfr[4];
        #pragma unroll
        for (int mt = 0; mt < 2; mt++) {
            const int mb = wm * 32 + mt * 16 + mrow;
            #pragma unroll
            for (int v = 0; v < 8; v++) {
                const int kk = (v < 4) ? (half * 8 + 2 * v)
                                       : (16 + half * 8 + 2 * (v - 4));
                afr[mt].u32[v] = *(const unsigned int*)(&lds_a[mb * LDSS + kk]);
            }
        }
        #pragma unroll
        for (int nt = 0; nt < 4; nt++) {
            const int nb = wn * 64 + nt * 16 + mrow;
            #pragma unroll
            for (int v = 0; v < 8; v++) {
                const int kk = (v < 4) ? (half * 8 + 2 * v)
                                       : (16 + half * 8 + 2 * (v - 4));
                bfr[nt].u32[v] = *(const unsigned int*)(&lds_b[nb * LDSS + kk]);
            }
        }
        #pragma unroll
        for (int mt = 0; mt < 2; mt++)
            #pragma unroll
            for (int nt = 0; nt < 4; nt++)
                acc[mt][nt] = __builtin_amdgcn_wmma_f32_16x16x32_bf16(
                    false, afr[mt].bf, false, bfr[nt].bf,
                    (short)0, acc[mt][nt], false, false);
        __syncthreads();
    }

    // ---- epilogue: D layout VGPR r -> row (r | r+8), N = lane%16 ----
    const int hadd = half * 8;
    #pragma unroll
    for (int mt = 0; mt < 2; mt++) {
        #pragma unroll
        for (int nt = 0; nt < 4; nt++) {
            const int gn = n0 + wn * 64 + nt * 16 + mrow;
            #pragma unroll
            for (int r = 0; r < 8; r++) {
                const int gm = m0 + wm * 32 + mt * 16 + r + hadd;
                const float v = acc[mt][nt][r];
                if constexpr (MODE == 0) {
                    if (gn < 2048) out0[(size_t)gm * 2048 + gn] = v;
                    else out1[(size_t)gm * 2048 + (gn - 2048)] =
                             v / (1.0f + __expf(-v));
                } else if constexpr (MODE == 1) {
                    if (gn < 96) out0[(size_t)gm * 96 + gn] = v;
                    if (gn < 64) out_bf[(size_t)gm * 64 + gn] = f2bf(v);
                } else if constexpr (MODE == 2) {
                    const float s = v + extra[gn];
                    const float sp = (s > 20.0f) ? s : __logf(1.0f + __expf(s));
                    out0[(size_t)gm * 2048 + gn] = sp;
                } else {
                    out0[(size_t)gm * 1024 + gn] =
                        v + extra[(size_t)gm * 1024 + gn];
                }
            }
        }
    }
}

// Depthwise causal conv (k=4) + bias + SiLU. One thread per (b,l,d).
__global__ __launch_bounds__(256)
void conv_silu_kernel(const float* __restrict__ x, const float* __restrict__ cw,
                      const float* __restrict__ cb,
                      float* __restrict__ xo, unsigned short* __restrict__ xo_bf) {
    const int idx = blockIdx.x * 256 + threadIdx.x;     // [0, 4096*2048)
    const int d   = idx & 2047;
    const int row = idx >> 11;                          // b*2048 + l
    const int l   = row & 2047;
    const float w0 = cw[d * 4 + 0], w1 = cw[d * 4 + 1];
    const float w2 = cw[d * 4 + 2], w3 = cw[d * 4 + 3];
    const float* base = x + (size_t)row * 2048 + d;
    float s = cb[d] + w3 * base[0];
    if (l >= 1) s += w2 * base[-2048];
    if (l >= 2) s += w1 * base[-4096];
    if (l >= 3) s += w0 * base[-6144];
    const float r = s / (1.0f + __expf(-s));
    xo[idx]    = r;
    xo_bf[idx] = f2bf(r);
}

// Selective scan: one lane per (b, d) channel, 16-state in registers.
// B_t/C_t shared per batch: staged in LDS in 64-timestep chunks.
__global__ __launch_bounds__(256)
void scan_kernel(const float* __restrict__ xc, const float* __restrict__ dt,
                 const float* __restrict__ xdbl, const float* __restrict__ zs,
                 const float* __restrict__ A_log, const float* __restrict__ Dp,
                 unsigned short* __restrict__ y_bf) {
    __shared__ float bc[64 * 32];   // [t_local][0..15]=B, [16..31]=C
    const int b = blockIdx.y;
    const int d = blockIdx.x * 256 + threadIdx.x;

    float A[16], h[16];
    #pragma unroll
    for (int n = 0; n < 16; n++) {
        A[n] = -__expf(A_log[d * 16 + n]);
        h[n] = 0.f;
    }
    const float Dd = Dp[d];

    for (int t0 = 0; t0 < 2048; t0 += 64) {
        __syncthreads();
        #pragma unroll
        for (int i = 0; i < 8; i++) {
            const int u  = threadIdx.x + i * 256;
            const int tl = u >> 5, c = u & 31;
            bc[u] = xdbl[(size_t)(b * 2048 + t0 + tl) * 96 + 64 + c];
        }
        __syncthreads();
        for (int tl = 0; tl < 64; tl++) {
            const size_t row = (size_t)(b * 2048 + t0 + tl);
            const float xv  = xc[row * 2048 + d];
            const float dtv = dt[row * 2048 + d];
            const float zv  = zs[row * 2048 + d];
            const float dx  = dtv * xv;
            const float* Bt = &bc[tl * 32];
            float y = 0.f;
            #pragma unroll
            for (int n = 0; n < 16; n++) {
                const float dA = __expf(dtv * A[n]);
                h[n] = dA * h[n] + dx * Bt[n];
                y += h[n] * Bt[16 + n];
            }
            y = (y + xv * Dd) * zv;
            y_bf[row * 2048 + d] = f2bf(y);
        }
    }
}

extern "C" void kernel_launch(void* const* d_in, const int* in_sizes, int n_in,
                              void* d_out, int out_size, void* d_ws, size_t ws_size,
                              hipStream_t stream) {
    const float* residual = (const float*)d_in[0];   // (2,2048,1024)
    const float* norm_w   = (const float*)d_in[1];   // (1024,)
    const float* W_in     = (const float*)d_in[2];   // (1024,4096)
    const float* conv_w   = (const float*)d_in[3];   // (2048,4)
    const float* conv_b   = (const float*)d_in[4];   // (2048,)
    const float* W_x      = (const float*)d_in[5];   // (2048,96)
    const float* W_dt     = (const float*)d_in[6];   // (64,2048)
    const float* b_dt     = (const float*)d_in[7];   // (2048,)
    const float* A_log    = (const float*)d_in[8];   // (2048,16)
    const float* Dp       = (const float*)d_in[9];   // (2048,)
    const float* W_out    = (const float*)d_in[10];  // (2048,1024)

    char* ws = (char*)d_ws;
    size_t off = 0;
    auto alloc = [&](size_t bytes) -> void* {
        void* p = ws + off;
        off += (bytes + 255) & ~(size_t)255;
        return p;
    };

    const size_t M = 4096;  // B*L
    unsigned short* hidden_bf = (unsigned short*)alloc(M * 1024 * 2);
    unsigned short* Win_bf    = (unsigned short*)alloc(1024 * 4096 * 2);
    unsigned short* Wx_bf     = (unsigned short*)alloc(2048 * 128 * 2);  // padded 96->128
    unsigned short* Wdt_bf    = (unsigned short*)alloc(64 * 2048 * 2);
    unsigned short* Wout_bf   = (unsigned short*)alloc(2048 * 1024 * 2);
    float*          x_raw     = (float*)alloc(M * 2048 * 4);
    float*          z_silu    = (float*)alloc(M * 2048 * 4);
    float*          x_conv    = (float*)alloc(M * 2048 * 4);
    unsigned short* xconv_bf  = (unsigned short*)alloc(M * 2048 * 2);
    float*          x_dbl     = (float*)alloc(M * 96 * 4);
    unsigned short* dtA_bf    = (unsigned short*)alloc(M * 64 * 2);
    float*          dt_buf    = (float*)alloc(M * 2048 * 4);
    unsigned short* y_bf      = (unsigned short*)alloc(M * 2048 * 2);

    // 1) weights fp32 -> bf16 (W_x zero-padded to 128 cols)
    cvt_bf16_kernel<<<(1024 * 4096) / 256, 256, 0, stream>>>(W_in,  Win_bf, 1024 * 4096);
    cvt_bf16_pad_kernel<<<(2048 * 128) / 256, 256, 0, stream>>>(W_x, Wx_bf, 2048, 96, 128);
    cvt_bf16_kernel<<<(64 * 2048) / 256, 256, 0, stream>>>(W_dt, Wdt_bf, 64 * 2048);
    cvt_bf16_kernel<<<(2048 * 1024) / 256, 256, 0, stream>>>(W_out, Wout_bf, 2048 * 1024);

    // 2) RMSNorm -> bf16 hidden
    rmsnorm_kernel<<<(int)M, 256, 0, stream>>>(residual, norm_w, hidden_bf);

    // 3) in-projection GEMM (4096x1024 @ 1024x4096), split x / silu(z)
    gemm_bf16_kernel<0, 4096, 4096, 1024><<<dim3(4096 / BN, M / BM), 256, 0, stream>>>(
        hidden_bf, Win_bf, x_raw, z_silu, nullptr, nullptr);

    // 4) depthwise causal conv + SiLU
    conv_silu_kernel<<<(int)(M * 2048 / 256), 256, 0, stream>>>(
        x_raw, conv_w, conv_b, x_conv, xconv_bf);

    // 5) x_dbl GEMM (4096x2048 @ 2048x128-padded); emit bf16 dt-rank slice
    gemm_bf16_kernel<1, 4096, 128, 2048><<<dim3(1, M / BM), 256, 0, stream>>>(
        xconv_bf, Wx_bf, x_dbl, nullptr, dtA_bf, nullptr);

    // 6) dt GEMM (4096x64 @ 64x2048) + bias + softplus
    gemm_bf16_kernel<2, 4096, 2048, 64><<<dim3(2048 / BN, M / BM), 256, 0, stream>>>(
        dtA_bf, Wdt_bf, dt_buf, nullptr, nullptr, b_dt);

    // 7) selective scan + D skip + z gating -> bf16 y
    scan_kernel<<<dim3(8, 2), 256, 0, stream>>>(
        x_conv, dt_buf, x_dbl, z_silu, A_log, Dp, y_bf);

    // 8) out-projection GEMM (4096x2048 @ 2048x1024) + residual add
    gemm_bf16_kernel<3, 4096, 1024, 2048><<<dim3(1024 / BN, M / BM), 256, 0, stream>>>(
        y_bf, Wout_bf, (float*)d_out, nullptr, nullptr, residual);
}
